// ReservoirExpert_42142219108813
// MI455X (gfx1250) — matrix operand order repaired and checked by reading, not compiled, not run on hardware
//
#include <hip/hip_runtime.h>
#include <hip/hip_bf16.h>

// ---------------------------------------------------------------------------
// Echo-state network on gfx1250.
//  kernel 1 (prep):    W_res f32 -> bf16 in ws, zero h0 + spin counters
//  kernel 2 (recur):   512 sequential steps, weight-stationary in LDS,
//                      v_wmma_f32_16x16x32_bf16, TDM tensor_load_to_lds for
//                      staging, per-batch-group global spin barrier.
//  kernel 3 (readout): h_final @ readout_w^T + b via WMMA.
// Assumptions (noted, unverifiable compile-only): 128 WGs co-resident (>=128
// WGPs), static LDS up to 320KB allowed, TDM D# per ISA sec 8 with lds_addr
// workgroup-relative, B-fragment layout per ISA 7.12.2 sparse-B table.
// ---------------------------------------------------------------------------

typedef __bf16 bf16_t;
typedef __attribute__((ext_vector_type(16))) __bf16 v16bf;
typedef __attribute__((ext_vector_type(8)))  __bf16 v8bf;
typedef __attribute__((ext_vector_type(8)))  float  v8f;
typedef __attribute__((ext_vector_type(4)))  unsigned int v4u;
typedef __attribute__((ext_vector_type(8)))  int v8i;
typedef __attribute__((ext_vector_type(4)))  int v4i;

#define RES_N    1024
#define BATCH_N  256
#define SEQ_N    512
#define DMODEL_N 512

#define GM 8                 // batch groups
#define GN 16                // reservoir-column groups
#define MB 32                // batch rows per WG   (BATCH_N/GM)
#define NB 64                // W_res rows per WG   (RES_N/GN)

// TDM pads 16B per 256 DWORDs (1024B) of data -> LDS row stride 2080 B.
#define ROW_B    2080u
#define W_LDS_B  (NB * ROW_B)            // 133120
#define H_OFF    W_LDS_B
#define H_LDS_B  (MB * ROW_B)            // 66560
#define XS_OFF   (W_LDS_B + H_LDS_B)
#define SMEM_B   (XS_OFF + MB * 4)       // 199808 bytes of LDS

// workspace layout (bytes)
#define WS_WRES  0u
#define WS_H0    (2u * 1024u * 1024u)
#define WS_H1    (WS_H0 + 512u * 1024u)
#define WS_CNT   (WS_H1 + 512u * 1024u)  // 8 counters; total < 3.1 MB

// ---------------------------------------------------------------------------
// TDM: 2D tile, rows x 1024 bf16, row-major (stride 1024), padded into LDS.
// D# group0: count=1, lds_addr, global_addr, type=2.
// D# group1: data_size=2B, pad_enable, pad_interval=256dw, pad_amount=4dw.
// Groups 2/3 zero (2D tile, tile_dim2=0). 6-arg builtin form (clang-23 lane):
// (g0 u32x4, g1 i32x8, g2 i32x4, g3 i32x4, i32x8, cpol).
// ---------------------------------------------------------------------------
__device__ __forceinline__ void tdm_load_tile_2d(unsigned lds_byte_off,
                                                 const bf16_t* gptr,
                                                 unsigned rows) {
  unsigned long long ga = (unsigned long long)gptr;
  v4u g0;
  g0[0] = 1u;                                            // count=1, user mode
  g0[1] = lds_byte_off;                                  // lds_addr
  g0[2] = (unsigned)ga;                                  // global_addr[31:0]
  g0[3] = ((unsigned)(ga >> 32) & 0x01FFFFFFu) | (2u << 30); // addr hi | type=2
  v8i g1;
  g1[0] = (int)0x07D10000u;        // data_size=1(2B) | pad_en | intv=7 | amt=3
  g1[1] = (int)(1024u << 16);      // tensor_dim0 = 1024
  g1[2] = (int)(rows << 16);       // tensor_dim1 = rows
  g1[3] = (int)(1024u << 16);      // tile_dim0 = 1024
  g1[4] = (int)rows;               // tile_dim1 = rows, tile_dim2 = 0
  g1[5] = 1024;                    // tensor_dim0_stride = 1024
  g1[6] = 0;
  g1[7] = 0;
  v4i z4 = 0;
  v8i z8 = 0;
  __builtin_amdgcn_tensor_load_to_lds(g0, g1, z4, z4, z8, 0);
}

// ---------------------------------------------------------------------------
// Kernel 1: prep (runs every launch: ws is poisoned / counters must reset)
// ---------------------------------------------------------------------------
__global__ void esn_prep_kernel(const float* __restrict__ wres,
                                unsigned char* __restrict__ ws) {
  const unsigned g = blockIdx.x * blockDim.x + threadIdx.x;
  bf16_t* wbf = (bf16_t*)(ws + WS_WRES);
  if (g < 1024u * 1024u) wbf[g] = (bf16_t)wres[g];
  unsigned short* h0 = (unsigned short*)(ws + WS_H0);
  if (g < 256u * 1024u) h0[g] = 0;                       // h0 = zeros (bf16)
  if (g < 8u) ((unsigned*)(ws + WS_CNT))[g] = 0u;        // spin counters
}

// ---------------------------------------------------------------------------
// Kernel 2: the recurrence. grid = 128 WGs (bi = bx>>4, bj = bx&15),
// block = 256 threads = 8 waves; wave w owns tile (mt=w>>2, nt=w&3).
// ---------------------------------------------------------------------------
__global__ void __launch_bounds__(256, 1)
esn_recur_kernel(const float* __restrict__ x,
                 const float* __restrict__ w_in,
                 unsigned char* __restrict__ ws) {
  __shared__ __align__(16) unsigned char smem[SMEM_B];

  const bf16_t* wres_bf = (const bf16_t*)(ws + WS_WRES);
  bf16_t* h0buf = (bf16_t*)(ws + WS_H0);
  bf16_t* h1buf = (bf16_t*)(ws + WS_H1);
  unsigned* cnt = (unsigned*)(ws + WS_CNT);

  const int tid = threadIdx.x;
  const int wv  = tid >> 5;
  const int ln  = tid & 15;          // lane % 16
  const int lh  = (tid >> 4) & 1;    // lane / 16 within wave
  const int bi  = blockIdx.x >> 4;   // batch group 0..7
  const int bj  = blockIdx.x & 15;   // column group 0..15
  const int mt  = wv >> 2;           // 0..1
  const int nt  = wv & 3;            // 0..3

  // pin this WG's W_res slice (rows bj*64 .. +64, all K) into LDS once
  if (wv == 0)
    tdm_load_tile_2d(0u, wres_bf + (size_t)bj * NB * RES_N, (unsigned)NB);

  const int ncol    = bj * NB + nt * 16 + ln;   // global reservoir column
  const float win_n = w_in[ncol];               // loaded once, lives in VGPR
  const unsigned aRow = (unsigned)H_OFF + (unsigned)(mt * 16 + ln) * ROW_B;
  const unsigned bRow = (unsigned)(nt * 16 + ln) * ROW_B;
  float* xs = (float*)(smem + XS_OFF);

  if (wv == 0) __builtin_amdgcn_s_wait_tensorcnt(0);
  __syncthreads();                               // W slice resident

  for (int t = 0; t < SEQ_N; ++t) {
    const bf16_t* hsrc = (t & 1) ? h1buf : h0buf;
    bf16_t*       hdst = (t & 1) ? h0buf : h1buf;

    // stage h_t (32 x 1024 bf16) into LDS via TDM; wave0 also fetches x[:,t]
    if (wv == 0) {
      tdm_load_tile_2d((unsigned)H_OFF, hsrc + (size_t)bi * MB * RES_N,
                       (unsigned)MB);
      const int lane = tid & 31;
      xs[lane] = x[(size_t)(bi * MB + lane) * SEQ_N + t];
      __builtin_amdgcn_s_wait_tensorcnt(0);
    }
    __syncthreads();

    v8f acc0 = 0.f, acc1 = 0.f;                  // 2 accs break WMMA RAW chain
#pragma unroll 8
    for (int kc = 0; kc < 32; ++kc) {
      const unsigned k0 = (unsigned)kc * 32u;
      const unsigned kb = 2u * k0 + ((k0 >= 512u) ? 16u : 0u); // skip mid-row pad
      // A (h tile, 16x32): two 16B chunks per lane (ISA 7.12.2)
      const v8bf a0 = *(const v8bf*)(smem + aRow + kb + 16u * lh);
      const v8bf a1 = *(const v8bf*)(smem + aRow + kb + 16u * lh + 32u);
      // B (W slice, 32x16): two 16B chunks per lane, same K order as A
      const v8bf b0 = *(const v8bf*)(smem + bRow + kb + 16u * lh);
      const v8bf b1 = *(const v8bf*)(smem + bRow + kb + 16u * lh + 32u);
      v16bf a = __builtin_shufflevector(a0, a1, 0,1,2,3,4,5,6,7,8,9,10,11,12,13,14,15);
      v16bf b = __builtin_shufflevector(b0, b1, 0,1,2,3,4,5,6,7,8,9,10,11,12,13,14,15);
      if (kc & 1)
        acc1 = __builtin_amdgcn_wmma_f32_16x16x32_bf16(false, a, false, b,
                                                       (short)0, acc1, false, false);
      else
        acc0 = __builtin_amdgcn_wmma_f32_16x16x32_bf16(false, a, false, b,
                                                       (short)0, acc0, false, false);
    }
    __syncthreads();                             // all h_lds reads complete

    // epilogue: + x_t*w_in, tanh, store bf16 h_{t+1} slice to global
#pragma unroll
    for (int r = 0; r < 8; ++r) {
      const int ml = mt * 16 + r + 8 * lh;       // C/D: M = r + 8*laneHi
      const float v = tanhf(acc0[r] + acc1[r] + xs[ml] * win_n);
      hdst[(size_t)(bi * MB + ml) * RES_N + ncol] = (bf16_t)v;
    }
    __threadfence();                             // flush stores to device scope
    __syncthreads();

    // batch-group spin barrier: monotonic counter, 16 WGs per group
    if (tid == 0) {
      __hip_atomic_fetch_add(&cnt[bi], 1u, __ATOMIC_RELEASE,
                             __HIP_MEMORY_SCOPE_AGENT);
      const unsigned tgt = (unsigned)(GN * (t + 1));
      while (__hip_atomic_load(&cnt[bi], __ATOMIC_ACQUIRE,
                               __HIP_MEMORY_SCOPE_AGENT) < tgt)
        __builtin_amdgcn_s_sleep(1);
    }
    __syncthreads();
  }
}

// ---------------------------------------------------------------------------
// Kernel 3: readout  out = h_final @ readout_w^T + b  (h_final bf16 in WS_H0,
// since step 511 writes buffer (511+1)&1 == 0). One 16x16 tile per wave.
// ---------------------------------------------------------------------------
__global__ void __launch_bounds__(256, 1)
esn_readout_kernel(const unsigned char* __restrict__ ws,
                   const float* __restrict__ rw,
                   const float* __restrict__ rb,
                   float* __restrict__ out) {
  const bf16_t* hfin = (const bf16_t*)(ws + WS_H0);
  const int tid = threadIdx.x;
  const int wv  = tid >> 5;
  const int ln  = tid & 15;
  const int lh  = (tid >> 4) & 1;
  const int tile = blockIdx.x * 8 + wv;   // 0..511
  const int mt = tile >> 5;               // 16 M-tiles (256 rows)
  const int nt = tile & 31;               // 32 N-tiles (512 cols)

  const int arow = mt * 16 + ln;
  const int ncol = nt * 16 + ln;
  const float bias = rb[ncol];

  v8f acc0 = 0.f, acc1 = 0.f;
#pragma unroll 4
  for (int kc = 0; kc < 32; ++kc) {
    const int k0 = kc * 32;
    const v8bf a0 = *(const v8bf*)(hfin + (size_t)arow * RES_N + k0 + lh * 16);
    const v8bf a1 = *(const v8bf*)(hfin + (size_t)arow * RES_N + k0 + lh * 16 + 8);
    v16bf a = __builtin_shufflevector(a0, a1, 0,1,2,3,4,5,6,7,8,9,10,11,12,13,14,15);
    const float* bp = rw + (size_t)ncol * RES_N + k0 + lh * 16;
    v16bf b;
#pragma unroll
    for (int e = 0; e < 16; ++e) b[e] = (bf16_t)bp[e];   // f32 -> bf16 cvt
    if (kc & 1)
      acc1 = __builtin_amdgcn_wmma_f32_16x16x32_bf16(false, a, false, b,
                                                     (short)0, acc1, false, false);
    else
      acc0 = __builtin_amdgcn_wmma_f32_16x16x32_bf16(false, a, false, b,
                                                     (short)0, acc0, false, false);
  }
#pragma unroll
  for (int r = 0; r < 8; ++r) {
    const int m = mt * 16 + r + 8 * lh;
    out[(size_t)m * DMODEL_N + ncol] = acc0[r] + acc1[r] + bias;
  }
}

// ---------------------------------------------------------------------------
extern "C" void kernel_launch(void* const* d_in, const int* in_sizes, int n_in,
                              void* d_out, int out_size, void* d_ws, size_t ws_size,
                              hipStream_t stream) {
  const float* x    = (const float*)d_in[0];   // (256, 512)
  const float* w_in = (const float*)d_in[1];   // (1024, 1)
  const float* wres = (const float*)d_in[2];   // (1024, 1024)
  const float* rw   = (const float*)d_in[3];   // (512, 1024)
  const float* rb   = (const float*)d_in[4];   // (512,)
  unsigned char* ws = (unsigned char*)d_ws;    // needs ~3.1 MB

  esn_prep_kernel<<<4096, 256, 0, stream>>>(wres, ws);
  esn_recur_kernel<<<GM * GN, 256, 0, stream>>>(x, w_in, ws);
  esn_readout_kernel<<<64, 256, 0, stream>>>(ws, rw, rb, (float*)d_out);
}
